// FullAttention_v4_47330539602002
// MI455X (gfx1250) — matrix operand order repaired and checked by reading, not compiled, not run on hardware
//
#include <hip/hip_runtime.h>
#include <hip/hip_bf16.h>
#include <math.h>

typedef __attribute__((ext_vector_type(16))) _Float16 v16h;
typedef __attribute__((ext_vector_type(8)))  _Float16 v8h;
typedef __attribute__((ext_vector_type(8)))  float    v8f;

#define B_ 8
#define H_ 8
#define L_ 1024
#define S_ 1024
#define E_ 64
#define D_ 64
#define NTOT  (B_*S_*H_*D_)   /* 4,194,304 */
#define HALFN (NTOT/2)

#define WAVES 4
#define ROWS_PER_WAVE 16
#define STILE 64

// ---------------- JAX threefry2x32 (key = [0,42]) ----------------
__device__ __forceinline__ void threefry2x32(uint32_t k0, uint32_t k1,
                                             uint32_t x0, uint32_t x1,
                                             uint32_t& y0, uint32_t& y1) {
  uint32_t ks0 = k0, ks1 = k1, ks2 = 0x1BD11BDAu ^ k0 ^ k1;
  x0 += ks0; x1 += ks1;
  const int rotA[4] = {13, 15, 26, 6};
  const int rotB[4] = {17, 29, 16, 24};
  #pragma unroll
  for (int g = 0; g < 5; ++g) {
    const int* r = (g & 1) ? rotB : rotA;
    #pragma unroll
    for (int j = 0; j < 4; ++j) {
      x0 += x1;
      x1 = (x1 << r[j]) | (x1 >> (32 - r[j]));
      x1 ^= x0;
    }
    uint32_t ka = (g == 0) ? ks1 : (g == 1) ? ks2 : (g == 2) ? ks0 : (g == 3) ? ks1 : ks2;
    uint32_t kb = (g == 0) ? ks2 : (g == 1) ? ks0 : (g == 2) ? ks1 : (g == 3) ? ks2 : ks0;
    x0 += ka; x1 += kb + (uint32_t)(g + 1);
  }
  y0 = x0; y1 = x1;
}

__device__ __forceinline__ float jax_normal42(uint32_t idx) {
  uint32_t p = (idx < HALFN) ? idx : idx - HALFN;
  uint32_t y0, y1;
  threefry2x32(0u, 42u, p, p + (uint32_t)HALFN, y0, y1);
  uint32_t bits = (idx < HALFN) ? y0 : y1;
  float f = __uint_as_float((bits >> 9) | 0x3f800000u) - 1.0f;  // [0,1)
  const float lo = -0.99999994f, hi = 1.0f;
  float u = fmaxf(lo, f * (hi - lo) + lo);
  return 1.41421356237f * erfinvf(u);
}

// ---------------- prep: Q/K -> f16 [B,H,L,E]; W=V-Z, Z -> f16 [B,H,D,S] ----------------
__global__ void prep_kernel(const float* __restrict__ q,
                            const float* __restrict__ k,
                            const float* __restrict__ v,
                            _Float16* __restrict__ qh,
                            _Float16* __restrict__ kh,
                            _Float16* __restrict__ wt,
                            _Float16* __restrict__ zt) {
  uint32_t idx = blockIdx.x * blockDim.x + threadIdx.x;
  if (idx >= NTOT) return;
  uint32_t b = idx >> 19, h = (idx >> 16) & 7u;
  {
    uint32_t e = idx & 63u, l = (idx >> 6) & 1023u;
    uint32_t src = ((b * L_ + l) * H_ + h) * E_ + e;
    qh[idx] = (_Float16)q[src];
    kh[idx] = (_Float16)k[src];
  }
  {
    uint32_t s = idx & 1023u, d = (idx >> 10) & 63u;
    uint32_t vsrc = ((b * S_ + s) * H_ + h) * D_ + d;
    float z = 0.1f * jax_normal42(vsrc);
    zt[idx] = (_Float16)z;
    wt[idx] = (_Float16)(v[vsrc] - z);
  }
}

// ---------------- 16-lane-row butterfly via v_permlane16_b32 (no LDS) ----------------
__device__ __forceinline__ float permx16(float v, uint32_t lo, uint32_t hi) {
  int x = __float_as_int(v);
  int y = __builtin_amdgcn_permlane16(x, x, (int)lo, (int)hi, false, false);
  return __int_as_float(y);
}
__device__ __forceinline__ float row_max16(float v) {
  v = fmaxf(v, permx16(v, 0x67452301u, 0xEFCDAB89u));  // xor 1
  v = fmaxf(v, permx16(v, 0x54761032u, 0xDCFE98BAu));  // xor 2
  v = fmaxf(v, permx16(v, 0x32107654u, 0xBA98FEDCu));  // xor 4
  v = fmaxf(v, permx16(v, 0xFEDCBA98u, 0x76543210u));  // xor 8
  return v;
}
__device__ __forceinline__ float row_sum16(float v) {
  v += permx16(v, 0x67452301u, 0xEFCDAB89u);
  v += permx16(v, 0x54761032u, 0xDCFE98BAu);
  v += permx16(v, 0x32107654u, 0xBA98FEDCu);
  v += permx16(v, 0xFEDCBA98u, 0x76543210u);
  return v;
}

// ---------------- fused flash attention with noise mixing ----------------
__global__ __launch_bounds__(WAVES * 32) void attn_kernel(
    const _Float16* __restrict__ qh, const _Float16* __restrict__ kh,
    const _Float16* __restrict__ wt, const _Float16* __restrict__ zt,
    const float* __restrict__ mask, float* __restrict__ out) {
  __shared__ _Float16 lds_p [WAVES][ROWS_PER_WAVE * STILE];
  __shared__ _Float16 lds_pp[WAVES][ROWS_PER_WAVE * STILE];

  const int lane = threadIdx.x & 31;
  const int wave = threadIdx.x >> 5;
  const int n    = lane & 15;
  const int hf   = lane >> 4;

  const int ltiles = L_ / (WAVES * ROWS_PER_WAVE);             // 16
  const int bh = blockIdx.x / ltiles;
  const int l0 = (blockIdx.x % ltiles) * (WAVES * ROWS_PER_WAVE) + wave * ROWS_PER_WAVE;

  const _Float16* Qbh = qh + (size_t)bh * L_ * E_;
  const _Float16* Kbh = kh + (size_t)bh * S_ * E_;
  const _Float16* Wbh = wt + (size_t)bh * D_ * S_;
  const _Float16* Zbh = zt + (size_t)bh * D_ * S_;

  // Resident Q A-fragments: e=0..31 and e=32..63
  v16h aq0, aq1;
  {
    const _Float16* qr = Qbh + (size_t)(l0 + n) * E_;
    v8h c0 = *(const v8h*)(qr + hf * 8);
    v8h c1 = *(const v8h*)(qr + 16 + hf * 8);
    v8h c2 = *(const v8h*)(qr + 32 + hf * 8);
    v8h c3 = *(const v8h*)(qr + 48 + hf * 8);
    #pragma unroll
    for (int j = 0; j < 8; ++j) {
      aq0[j] = c0[j]; aq0[j + 8] = c1[j];
      aq1[j] = c2[j]; aq1[j + 8] = c3[j];
    }
  }

  v8f accs[4] = {};
  float rmax[8], rsum[8];
  #pragma unroll
  for (int i = 0; i < 8; ++i) { rmax[i] = -1e30f; rsum[i] = 0.0f; }

  const float scale = 0.125f;   // 1/sqrt(64)

  #pragma unroll 1
  for (int s0 = 0; s0 < S_; s0 += STILE) {
    // ---- QK^T: four 16-column score fragments (8 WMMA)
    v8f c[4];
    #pragma unroll
    for (int j = 0; j < 4; ++j) {
      const _Float16* kr = Kbh + (size_t)(s0 + j * 16 + n) * E_;
      v8h k0 = *(const v8h*)(kr + hf * 8);
      v8h k1 = *(const v8h*)(kr + 16 + hf * 8);
      v8h k2 = *(const v8h*)(kr + 32 + hf * 8);
      v8h k3 = *(const v8h*)(kr + 48 + hf * 8);
      v16h bk0, bk1;
      #pragma unroll
      for (int t = 0; t < 8; ++t) {
        bk0[t] = k0[t]; bk0[t + 8] = k1[t];
        bk1[t] = k2[t]; bk1[t + 8] = k3[t];
      }
      v8f cc = {};
      cc = __builtin_amdgcn_wmma_f32_16x16x32_f16(false, aq0, false, bk0, (short)0, cc, false, false);
      cc = __builtin_amdgcn_wmma_f32_16x16x32_f16(false, aq1, false, bk1, (short)0, cc, false, false);
      c[j] = cc;
    }

    // ---- online softmax: all-VALU row reductions + mask + LDS staging
    float alpha[8];
    #pragma unroll
    for (int i = 0; i < 8; ++i) {
      float v0 = c[0][i] * scale, v1 = c[1][i] * scale;
      float v2 = c[2][i] * scale, v3 = c[3][i] * scale;
      float mx = fmaxf(fmaxf(v0, v1), fmaxf(v2, v3));
      mx = row_max16(mx);
      float nm = fmaxf(rmax[i], mx);
      float a  = __expf(rmax[i] - nm);
      float e0 = __expf(v0 - nm), e1 = __expf(v1 - nm);
      float e2 = __expf(v2 - nm), e3 = __expf(v3 - nm);
      float ts = row_sum16((e0 + e1) + (e2 + e3));
      rsum[i] = rsum[i] * a + ts;
      rmax[i] = nm; alpha[i] = a;

      int row = i + 8 * hf;
      size_t mrow = (size_t)(l0 + row) * S_ + s0;
      float P0 = fminf(fmaxf(mask[mrow + n],      0.0f), 1.0f);
      float P1 = fminf(fmaxf(mask[mrow + 16 + n], 0.0f), 1.0f);
      float P2 = fminf(fmaxf(mask[mrow + 32 + n], 0.0f), 1.0f);
      float P3 = fminf(fmaxf(mask[mrow + 48 + n], 0.0f), 1.0f);
      _Float16* dp  = &lds_p [wave][row * STILE + n];
      _Float16* dpp = &lds_pp[wave][row * STILE + n];
      dp[0]  = (_Float16)e0;  dp[16]  = (_Float16)e1;
      dp[32] = (_Float16)e2;  dp[48]  = (_Float16)e3;
      dpp[0]  = (_Float16)(e0 * P0);  dpp[16] = (_Float16)(e1 * P1);
      dpp[32] = (_Float16)(e2 * P2);  dpp[48] = (_Float16)(e3 * P3);
    }
    #pragma unroll
    for (int kk = 0; kk < 4; ++kk)
      #pragma unroll
      for (int i = 0; i < 8; ++i) accs[kk][i] *= alpha[i];

    __syncthreads();

    // ---- reload p / p*P as A-layout fragments (s 0..31 and 32..63)
    v16h ap0, ap1, app0, app1;
    {
      const _Float16* bp  = &lds_p [wave][n * STILE];
      const _Float16* bpp = &lds_pp[wave][n * STILE];
      v8h a0 = *(const v8h*)(bp  + hf * 8);
      v8h a1 = *(const v8h*)(bp  + 16 + hf * 8);
      v8h a2 = *(const v8h*)(bp  + 32 + hf * 8);
      v8h a3 = *(const v8h*)(bp  + 48 + hf * 8);
      v8h b0 = *(const v8h*)(bpp + hf * 8);
      v8h b1 = *(const v8h*)(bpp + 16 + hf * 8);
      v8h b2 = *(const v8h*)(bpp + 32 + hf * 8);
      v8h b3 = *(const v8h*)(bpp + 48 + hf * 8);
      #pragma unroll
      for (int t = 0; t < 8; ++t) {
        ap0[t] = a0[t];  ap0[t + 8] = a1[t];
        ap1[t] = a2[t];  ap1[t + 8] = a3[t];
        app0[t] = b0[t]; app0[t + 8] = b1[t];
        app1[t] = b2[t]; app1[t + 8] = b3[t];
      }
    }
    __syncthreads();

    // ---- PV: acc += (p.P) @ (V-Z) + p @ Z over 4 d-chunks (16 WMMA)
    #pragma unroll
    for (int kk = 0; kk < 4; ++kk) {
      const _Float16* wr = Wbh + (size_t)(kk * 16 + n) * S_ + s0 + hf * 8;
      const _Float16* zr = Zbh + (size_t)(kk * 16 + n) * S_ + s0 + hf * 8;
      v8h w0 = *(const v8h*)wr,        w1 = *(const v8h*)(wr + 16);
      v8h w2 = *(const v8h*)(wr + 32), w3 = *(const v8h*)(wr + 48);
      v8h z0 = *(const v8h*)zr,        z1 = *(const v8h*)(zr + 16);
      v8h z2 = *(const v8h*)(zr + 32), z3 = *(const v8h*)(zr + 48);
      v16h bw0, bw1, bz0, bz1;
      #pragma unroll
      for (int t = 0; t < 8; ++t) {
        bw0[t] = w0[t]; bw0[t + 8] = w1[t];
        bw1[t] = w2[t]; bw1[t + 8] = w3[t];
        bz0[t] = z0[t]; bz0[t + 8] = z1[t];
        bz1[t] = z2[t]; bz1[t + 8] = z3[t];
      }
      accs[kk] = __builtin_amdgcn_wmma_f32_16x16x32_f16(false, app0, false, bw0, (short)0, accs[kk], false, false);
      accs[kk] = __builtin_amdgcn_wmma_f32_16x16x32_f16(false, app1, false, bw1, (short)0, accs[kk], false, false);
      accs[kk] = __builtin_amdgcn_wmma_f32_16x16x32_f16(false, ap0,  false, bz0, (short)0, accs[kk], false, false);
      accs[kk] = __builtin_amdgcn_wmma_f32_16x16x32_f16(false, ap1,  false, bz1, (short)0, accs[kk], false, false);
    }
  }

  // ---- epilogue: out[b,l,h,d] = acc / rowsum
  const int b = bh >> 3, h = bh & 7;
  #pragma unroll
  for (int kk = 0; kk < 4; ++kk) {
    #pragma unroll
    for (int i = 0; i < 8; ++i) {
      int gl = l0 + i + 8 * hf;
      int d  = kk * 16 + n;
      out[(((size_t)b * L_ + gl) * H_ + h) * D_ + d] = accs[kk][i] / rsum[i];
    }
  }
}

extern "C" void kernel_launch(void* const* d_in, const int* in_sizes, int n_in,
                              void* d_out, int out_size, void* d_ws, size_t ws_size,
                              hipStream_t stream) {
  const float* q    = (const float*)d_in[0];
  const float* k    = (const float*)d_in[1];
  const float* v    = (const float*)d_in[2];
  const float* mask = (const float*)d_in[3];
  float* out = (float*)d_out;

  _Float16* qh = (_Float16*)d_ws;
  _Float16* kh = qh + NTOT;
  _Float16* wt = kh + NTOT;
  _Float16* zt = wt + NTOT;   // 32 MB workspace

  prep_kernel<<<NTOT / 256, 256, 0, stream>>>(q, k, v, qh, kh, wt, zt);
  attn_kernel<<<(B_ * H_) * (L_ / (WAVES * ROWS_PER_WAVE)), WAVES * 32, 0, stream>>>(
      qh, kh, wt, zt, mask, out);
}
